// Informer_42777874268776
// MI455X (gfx1250) — matrix operand order repaired and checked by reading, not compiled, not run on hardware
//
#include <hip/hip_runtime.h>
#include <math.h>

typedef __attribute__((ext_vector_type(16))) _Float16 v16h;
typedef __attribute__((ext_vector_type(8)))  _Float16 v8h;
typedef __attribute__((ext_vector_type(8)))  float    v8f;
typedef __attribute__((ext_vector_type(2)))  float    f2;

#define BSZ   8
#define NH    8
#define DHEAD 32
#define DM    256

// ---------------------------------------------------------------------------
// Batched WMMA GEMM:  C = alpha * A(MxK) * B(KxN) [+ bias] [GELU] [+= C]
//   block = 128 threads = 4 waves; waves split M (64 rows), block tile 64x32.
//   B slab (32k x 32n) staged in LDS as f16 [n][k]; each wave: 1 A frag,
//   2 B frags, 2 WMMAs per K-step. OOB rows/cols are CLAMPED (not zeroed):
//   row m of A only affects row m of C, col n of B only affects col n of C,
//   both guarded at the epilogue store -> no in-loop predication. Only the
//   K%32 tail zero-fills.
//   BMODE (compile-time): 0: B[k,n] = B[k*ldb + n]         (row-major weights)
//                         1: B[k,n] = B[n*ldb + k]         (transposed, contiguous k)
//                         2: B[k,n] = B[n*ldb + k*3]       (conv weights OIH, stride 3)
//   hA/hB/hC: per-head batch addressing off = (z/NH)*stride + (z%NH)*32.
//   rmod/rshift: circular row remap of A. flags: bit0 GELU, bit1 accumulate.
// ---------------------------------------------------------------------------
template <int BMODE>
__global__ void wmma_gemm_kernel(const float* __restrict__ A, long lda, long sA, int hA,
                                 const float* __restrict__ Bm, long ldb, long sB, int hB,
                                 const float* __restrict__ bias,
                                 float* __restrict__ C, long ldc, long sC, int hC,
                                 int M, int N, int K, float alpha, int flags,
                                 int rmod, int rshift)
{
  const int n0 = blockIdx.y * 32;
  if (n0 >= N || (int)(blockIdx.x * 64) >= M) return;   // block-uniform exit

  const int tid  = threadIdx.x;
  const int wave = tid >> 5;
  const int lane = tid & 31;
  const int m0   = blockIdx.x * 64 + wave * 16;

  const int z  = blockIdx.z;
  const int bb = z / NH, hh = z % NH;
  const float* Ap = hA ? (A  + (long)bb * sA + hh * DHEAD) : (A  + (long)z * sA);
  const float* Bp = hB ? (Bm + (long)bb * sB + hh * DHEAD) : (Bm + (long)z * sB);
  float*       Cp = hC ? (C  + (long)bb * sC + hh * DHEAD) : (C  + (long)z * sC);

  const int rlo    = lane & 15;
  const int khalfA = (lane >> 4) * 16;     // A frag: lanes16-31 hold K+16..31
  const int khalfB = (lane >> 4) * 16;     // B frag: lanes16-31 hold K+16..31

  const int arow = m0 + rlo;
  int asrc = arow;
  if (rmod > 0) { asrc = (arow + rshift) % rmod; if (asrc < 0) asrc += rmod; }
  if (arow >= M) asrc = 0;                 // clamp: garbage row never stored
  const float* Arow = Ap + (long)asrc * lda;

  // B slab [n][k]: n = 0..31 (tile cols), k = 0..31
  __shared__ _Float16 Bsh[32 * 32];

  // coop mapping: each thread owns col cn, 8 consecutive k at ckb
  const int cn  = tid & 31;
  const int ckb = (tid >> 5) * 8;
  int gn = n0 + cn; if (gn >= N) gn = N - 1;   // clamp: garbage col never stored

  // per-thread B base pointer at k = ckb (advance by 32 k per iteration)
  const float* pbt;
  long pbstep;
  if (BMODE == 0) { pbt = Bp + (long)ckb * ldb + gn;        pbstep = 32 * ldb; }
  else if (BMODE == 1) { pbt = Bp + (long)gn * ldb + ckb;   pbstep = 32; }
  else { pbt = Bp + (long)gn * ldb + (long)ckb * 3;         pbstep = 96; }

  v8f acc0 = {}, acc1 = {};
  const int kmain = K & ~31;

  for (int k0 = 0; k0 < kmain; k0 += 32) {
    __syncthreads();
    v8h hv;
    if (BMODE == 0) {
#pragma unroll
      for (int i = 0; i < 8; ++i) hv[i] = (_Float16)pbt[(long)i * ldb];
    } else if (BMODE == 1) {
#pragma unroll
      for (int i = 0; i < 4; ++i) {
        f2 y = ((const f2*)pbt)[i];
        hv[2 * i] = (_Float16)y.x; hv[2 * i + 1] = (_Float16)y.y;
      }
    } else {
#pragma unroll
      for (int i = 0; i < 8; ++i) hv[i] = (_Float16)pbt[i * 3];
    }
    pbt += pbstep;
    *(v8h*)&Bsh[cn * 32 + ckb] = hv;
    __syncthreads();

    v16h a;
    const float* pa = Arow + k0 + khalfA;
#pragma unroll
    for (int i = 0; i < 4; ++i) {
      f2 x0 = ((const f2*)pa)[i];
      f2 x1 = ((const f2*)(pa + 8))[i];
      a[2 * i]     = (_Float16)x0.x; a[2 * i + 1]     = (_Float16)x0.y;
      a[8 + 2 * i] = (_Float16)x1.x; a[8 + 2 * i + 1] = (_Float16)x1.y;
    }

    const v8h* f0 = (const v8h*)&Bsh[rlo * 32 + khalfB];
    v16h b0 = __builtin_shufflevector(f0[0], f0[1], 0, 1, 2, 3, 4, 5, 6, 7,
                                      8, 9, 10, 11, 12, 13, 14, 15);
    const v8h* f1 = (const v8h*)&Bsh[(16 + rlo) * 32 + khalfB];
    v16h b1 = __builtin_shufflevector(f1[0], f1[1], 0, 1, 2, 3, 4, 5, 6, 7,
                                      8, 9, 10, 11, 12, 13, 14, 15);

    acc0 = __builtin_amdgcn_wmma_f32_16x16x32_f16(false, a, false, b0,
                                                  (short)0, acc0, false, false);
    acc1 = __builtin_amdgcn_wmma_f32_16x16x32_f16(false, a, false, b1,
                                                  (short)0, acc1, false, false);
  }

  if (kmain < K) {                          // single guarded tail (K % 32)
    __syncthreads();
    v8h hv;
#pragma unroll
    for (int i = 0; i < 8; ++i) {
      int kk = kmain + ckb + i;
      float bv = 0.0f;
      if (kk < K) {
        if (BMODE == 0)      bv = Bp[(long)kk * ldb + gn];
        else if (BMODE == 1) bv = Bp[(long)gn * ldb + kk];
        else                 bv = Bp[(long)gn * ldb + (long)kk * 3];
      }
      hv[i] = (_Float16)bv;
    }
    *(v8h*)&Bsh[cn * 32 + ckb] = hv;
    __syncthreads();

    v16h a;
#pragma unroll
    for (int e = 0; e < 16; ++e) {
      int kk = kmain + khalfA + e;
      float av = (kk < K) ? Arow[kk] : 0.0f;
      a[e] = (_Float16)av;
    }
    const v8h* f0 = (const v8h*)&Bsh[rlo * 32 + khalfB];
    v16h b0 = __builtin_shufflevector(f0[0], f0[1], 0, 1, 2, 3, 4, 5, 6, 7,
                                      8, 9, 10, 11, 12, 13, 14, 15);
    const v8h* f1 = (const v8h*)&Bsh[(16 + rlo) * 32 + khalfB];
    v16h b1 = __builtin_shufflevector(f1[0], f1[1], 0, 1, 2, 3, 4, 5, 6, 7,
                                      8, 9, 10, 11, 12, 13, 14, 15);
    acc0 = __builtin_amdgcn_wmma_f32_16x16x32_f16(false, a, false, b0,
                                                  (short)0, acc0, false, false);
    acc1 = __builtin_amdgcn_wmma_f32_16x16x32_f16(false, a, false, b1,
                                                  (short)0, acc1, false, false);
  }

  // epilogue: two 16-wide column sets
  const int rbase = m0 + ((lane >> 4) * 8);
#pragma unroll
  for (int half = 0; half < 2; ++half) {
    int ccol = n0 + half * 16 + rlo;
    if (ccol < N) {
      float bcst = bias ? bias[ccol] : 0.0f;
#pragma unroll
      for (int e = 0; e < 8; ++e) {
        int crow = rbase + e;
        if (crow < M) {
          float v2 = (half ? acc1[e] : acc0[e]) * alpha + bcst;
          if (flags & 1) v2 = 0.5f * v2 * (1.0f + erff(v2 * 0.70710678118f));
          long ci = (long)crow * ldc + ccol;
          if (flags & 2) v2 += Cp[ci];
          Cp[ci] = v2;
        }
      }
    }
  }
}

// --------------------------- embedding (conv k=3 circular pad 1 + PE) ------
__global__ void embed_kernel(const float* __restrict__ x, const float* __restrict__ w,
                             float* __restrict__ out, int L)
{
  int d = threadIdx.x, l = blockIdx.x, b = blockIdx.y;
  float acc = 0.0f;
  for (int t = 0; t < 3; ++t) {
    int src = l + t - 1; if (src < 0) src += L; if (src >= L) src -= L;
    const float* xr = x + ((long)b * L + src) * 10;
    for (int c = 0; c < 10; ++c) acc += xr[c] * w[d * 30 + c * 3 + t];
  }
  int j = d >> 1;
  float div = expf(-logf(10000.0f) * (2.0f * (float)j) / 256.0f);
  float ang = (float)l * div;
  acc += (d & 1) ? cosf(ang) : sinf(ang);
  out[((long)b * L + l) * DM + d] = acc;
}

// --------------------------- LayerNorm (optional residual) -----------------
__global__ void layernorm_kernel(const float* __restrict__ x, const float* __restrict__ res,
                                 const float* __restrict__ g, const float* __restrict__ bb,
                                 float* __restrict__ out, int rows)
{
  int row = blockIdx.x, t = threadIdx.x;
  __shared__ float red[256];
  float v = x[(long)row * DM + t];
  if (res) v += res[(long)row * DM + t];
  red[t] = v; __syncthreads();
  for (int o = 128; o > 0; o >>= 1) { if (t < o) red[t] += red[t + o]; __syncthreads(); }
  float mean = red[0] / 256.0f; __syncthreads();
  float d = v - mean;
  red[t] = d * d; __syncthreads();
  for (int o = 128; o > 0; o >>= 1) { if (t < o) red[t] += red[t + o]; __syncthreads(); }
  float var = red[0] / 256.0f;
  out[(long)row * DM + t] = d * rsqrtf(var + 1e-5f) * g[t] + bb[t];
}

// --------------------------- distil: BN stats / apply+ELU / maxpool --------
__global__ void bn_stats_kernel(const float* __restrict__ y, int rows,
                                float* __restrict__ mean, float* __restrict__ var)
{
  int d = blockIdx.x, t = threadIdx.x;
  float s = 0.0f, s2 = 0.0f;
  for (int r = t; r < rows; r += 256) {
    float v = y[(long)r * DM + d]; s += v; s2 += v * v;
  }
  __shared__ float sm[256], sm2[256];
  sm[t] = s; sm2[t] = s2; __syncthreads();
  for (int o = 128; o > 0; o >>= 1) {
    if (t < o) { sm[t] += sm[t + o]; sm2[t] += sm2[t + o]; } __syncthreads();
  }
  if (t == 0) { float m = sm[0] / rows; mean[d] = m; var[d] = sm2[0] / rows - m * m; }
}

__global__ void bn_elu_kernel(float* __restrict__ y, const float* __restrict__ mean,
                              const float* __restrict__ var, const float* __restrict__ g,
                              const float* __restrict__ b, long n)
{
  long i = (long)blockIdx.x * blockDim.x + threadIdx.x;
  if (i >= n) return;
  int d = (int)(i & (DM - 1));
  float v = (y[i] - mean[d]) * rsqrtf(var[d] + 1e-5f) * g[d] + b[d];
  y[i] = v > 0.0f ? v : (expf(v) - 1.0f);
}

__global__ void maxpool_kernel(const float* __restrict__ y, float* __restrict__ out,
                               int Lc, int Lp)
{
  int d = threadIdx.x, i = blockIdx.x, b = blockIdx.y;
  int s = i * 2 - 1;
  float m = -1e30f;
  for (int t = 0; t < 3; ++t) {
    int idx = s + t;
    if (idx >= 0 && idx < Lc) m = fmaxf(m, y[((long)b * Lc + idx) * DM + d]);
  }
  out[((long)b * Lp + i) * DM + d] = m;
}

// --------------------------- ProbSparse attention pieces -------------------
__global__ void prob_m_kernel(const float* __restrict__ q, const float* __restrict__ k,
                              float* __restrict__ Mout, int L, int U, int seed)
{
  long idx = (long)blockIdx.x * blockDim.x + threadIdx.x;
  long tot = (long)BSZ * NH * L;
  if (idx >= tot) return;
  int l = (int)(idx % L);
  int bh = (int)(idx / L);
  int b = bh / NH, h = bh % NH;
  const float* qr = q + ((long)b * L + l) * DM + h * DHEAD;
  float mx = -1e30f, sm = 0.0f;
  for (int j = 0; j < U; ++j) {
    unsigned hsh = (unsigned)(l * 131071) ^ (unsigned)(j * 2654435761u)
                 ^ (unsigned)(seed * 97531);
    hsh ^= hsh >> 13; hsh *= 0x5bd1e995u; hsh ^= hsh >> 15;
    int s = (int)(hsh % (unsigned)L);
    const float* kr = k + ((long)b * L + s) * DM + h * DHEAD;
    float d = 0.0f;
    for (int c = 0; c < DHEAD; ++c) d += qr[c] * kr[c];
    mx = fmaxf(mx, d); sm += d;
  }
  Mout[idx] = mx - sm / (float)L;
}

__global__ void topk_kernel(const float* __restrict__ Mv, int* __restrict__ top,
                            int L, int u)
{
  int bh = blockIdx.x, t = threadIdx.x;
  extern __shared__ float smv[];
  __shared__ float redv[256];
  __shared__ int   redi[256];
  for (int i = t; i < L; i += 256) smv[i] = Mv[(long)bh * L + i];
  __syncthreads();
  for (int it = 0; it < u; ++it) {
    float bv = -1e30f; int bi = 0;
    for (int i = t; i < L; i += 256) { float v = smv[i]; if (v > bv) { bv = v; bi = i; } }
    redv[t] = bv; redi[t] = bi; __syncthreads();
    for (int o = 128; o > 0; o >>= 1) {
      if (t < o) {
        if (redv[t + o] > redv[t] ||
            (redv[t + o] == redv[t] && redi[t + o] < redi[t])) {
          redv[t] = redv[t + o]; redi[t] = redi[t + o];
        }
      }
      __syncthreads();
    }
    if (t == 0) { top[bh * u + it] = redi[0]; smv[redi[0]] = -1e30f; }
    __syncthreads();
  }
}

__global__ void ctx_mean_kernel(const float* __restrict__ v, float* __restrict__ ctx, int L)
{
  int b = blockIdx.x, c = threadIdx.x;
  float s = 0.0f;
  for (int l = 0; l < L; ++l) s += v[((long)b * L + l) * DM + c];
  s /= (float)L;
  for (int l = 0; l < L; ++l) ctx[((long)b * L + l) * DM + c] = s;
}

__global__ void ctx_cumsum_kernel(const float* __restrict__ v, float* __restrict__ ctx, int L)
{
  int b = blockIdx.x, c = threadIdx.x;
  float s = 0.0f;
  for (int l = 0; l < L; ++l) {
    s += v[((long)b * L + l) * DM + c];
    ctx[((long)b * L + l) * DM + c] = s;
  }
}

__global__ void prob_update_kernel(const float* __restrict__ q, const float* __restrict__ k,
                                   const float* __restrict__ v, const int* __restrict__ top,
                                   float* __restrict__ ctx, int L, int u, int maskflag)
{
  int bh = blockIdx.x, ui = blockIdx.y;
  int b = bh / NH, h = bh % NH, t = threadIdx.x;
  int row = top[bh * u + ui];
  extern __shared__ float sc[];
  const float* qr = q + ((long)b * L + row) * DM + h * DHEAD;
  for (int s = t; s < L; s += 256) {
    const float* kr = k + ((long)b * L + s) * DM + h * DHEAD;
    float d = 0.0f;
    for (int c = 0; c < DHEAD; ++c) d += qr[c] * kr[c];
    d *= 0.176776695f;                      // 1/sqrt(32)
    if (maskflag && s > row) d = -1e30f;
    sc[s] = d;
  }
  __syncthreads();
  __shared__ float red[256];
  float mx = -1e30f;
  for (int s = t; s < L; s += 256) mx = fmaxf(mx, sc[s]);
  red[t] = mx; __syncthreads();
  for (int o = 128; o > 0; o >>= 1) { if (t < o) red[t] = fmaxf(red[t], red[t + o]); __syncthreads(); }
  mx = red[0]; __syncthreads();
  float sm = 0.0f;
  for (int s = t; s < L; s += 256) { float e = __expf(sc[s] - mx); sc[s] = e; sm += e; }
  red[t] = sm; __syncthreads();
  for (int o = 128; o > 0; o >>= 1) { if (t < o) red[t] += red[t + o]; __syncthreads(); }
  float inv = 1.0f / red[0]; __syncthreads();
  if (t < DHEAD) {
    float o2 = 0.0f;
    for (int s = 0; s < L; ++s) o2 += sc[s] * v[((long)b * L + s) * DM + h * DHEAD + t];
    ctx[((long)b * L + row) * DM + h * DHEAD + t] = o2 * inv;
  }
}

// out.transpose(0,2,1,3).reshape(B,L,256) re-index (Informer "mix")
__global__ void mix_transpose_kernel(const float* __restrict__ ctx, float* __restrict__ out,
                                     int L, long n)
{
  long i = (long)blockIdx.x * blockDim.x + threadIdx.x;
  if (i >= n) return;
  int c = (int)(i & (DM - 1));
  long bl = i >> 8;
  int l = (int)(bl % L);
  int b = (int)(bl / L);
  long f = (long)l * DM + c;
  long hl = (long)DHEAD * L;
  int h  = (int)(f / hl);
  long r = f % hl;
  int l2 = (int)(r / DHEAD);
  int d  = (int)(r % DHEAD);
  out[i] = ctx[((long)b * L + l2) * DM + h * DHEAD + d];
}

__global__ void softmax_rows_kernel(float* __restrict__ s, long rows, int n)
{
  long r = (long)blockIdx.x * blockDim.x + threadIdx.x;
  if (r >= rows) return;
  float* p = s + r * (long)n;
  float mx = -1e30f;
  for (int i = 0; i < n; ++i) mx = fmaxf(mx, p[i]);
  float sm = 0.0f;
  for (int i = 0; i < n; ++i) { float e = __expf(p[i] - mx); p[i] = e; sm += e; }
  float inv = 1.0f / sm;
  for (int i = 0; i < n; ++i) p[i] *= inv;
}

// ===========================================================================
//                               HOST SIDE
// ===========================================================================
struct AttnP { const float *qw,*qb,*kw,*kb,*vw,*vb,*ow,*ob; };
struct EncLP { AttnP a; const float *g1,*b1,*g2,*b2,*f1w,*f1b,*f2w,*f2b; };
struct ConvP { const float *w,*b,*bg,*bb; };
struct DecLP { AttnP s, c; const float *g1,*b1,*g2,*b2,*g3,*b3,*f1w,*f1b,*f2w,*f2b; };

static void gemm(hipStream_t st, int bmode,
                 const float* A, long lda, long sA, int hA,
                 const float* Bm, long ldb, long sB, int hB,
                 const float* bias, float* C, long ldc, long sC, int hC,
                 int M, int N, int K, float alpha, int flags, int batch,
                 int rmod = 0, int rshift = 0)
{
  dim3 grid((unsigned)((M + 63) / 64), (unsigned)((N + 31) / 32), (unsigned)batch);
  dim3 blk(128, 1, 1);
  if (bmode == 0)
    wmma_gemm_kernel<0><<<grid, blk, 0, st>>>(A, lda, sA, hA, Bm, ldb, sB, hB,
        bias, C, ldc, sC, hC, M, N, K, alpha, flags, rmod, rshift);
  else if (bmode == 1)
    wmma_gemm_kernel<1><<<grid, blk, 0, st>>>(A, lda, sA, hA, Bm, ldb, sB, hB,
        bias, C, ldc, sC, hC, M, N, K, alpha, flags, rmod, rshift);
  else
    wmma_gemm_kernel<2><<<grid, blk, 0, st>>>(A, lda, sA, hA, Bm, ldb, sB, hB,
        bias, C, ldc, sC, hC, M, N, K, alpha, flags, rmod, rshift);
}

static void linear(hipStream_t st, const float* X, const float* W, const float* bias,
                   float* Y, int rows, int Kin, int Nout, int flags = 0)
{
  gemm(st, 0, X, Kin, 0, 0, W, Nout, 0, 0, bias, Y, Nout, 0, 0,
       rows, Nout, Kin, 1.0f, flags, 1);
}

static void ln(hipStream_t st, const float* x, const float* res, const float* g,
               const float* b, float* out, int rows)
{
  layernorm_kernel<<<rows, 256, 0, st>>>(x, res, g, b, out, rows);
}

static void attn_prob(hipStream_t st, float* X, int L, const AttnP& p, int mask, int mix,
                      float* Q, float* K, float* V, float* CTX, float* T1, float* AO,
                      float* MB, int* TOP, int seed)
{
  int rows = BSZ * L;
  linear(st, X, p.qw, p.qb, Q, rows, DM, DM);
  linear(st, X, p.kw, p.kb, K, rows, DM, DM);
  linear(st, X, p.vw, p.vb, V, rows, DM, DM);
  int U = (int)(5.0 * ceil(log((double)L))); if (U > L) U = L;
  int u = U;
  long tot = (long)BSZ * NH * L;
  prob_m_kernel<<<(unsigned)((tot + 255) / 256), 256, 0, st>>>(Q, K, MB, L, U, seed);
  topk_kernel<<<BSZ * NH, 256, (size_t)L * sizeof(float), st>>>(MB, TOP, L, u);
  if (mask) ctx_cumsum_kernel<<<BSZ, 256, 0, st>>>(V, CTX, L);
  else      ctx_mean_kernel<<<BSZ, 256, 0, st>>>(V, CTX, L);
  prob_update_kernel<<<dim3(BSZ * NH, u), 256, (size_t)L * sizeof(float), st>>>(
      Q, K, V, TOP, CTX, L, u, mask);
  const float* ain = CTX;
  if (mix) {
    long n = (long)rows * DM;
    mix_transpose_kernel<<<(unsigned)((n + 255) / 256), 256, 0, st>>>(CTX, T1, L, n);
    ain = T1;
  }
  linear(st, ain, p.ow, p.ob, AO, rows, DM, DM);
}

static void attn_full(hipStream_t st, const float* Xq, int Lq, const float* Xkv, int Lkv,
                      const AttnP& p, float* Q, float* K, float* V, float* SC,
                      float* T2, float* AO)
{
  linear(st, Xq,  p.qw, p.qb, Q, BSZ * Lq,  DM, DM);
  linear(st, Xkv, p.kw, p.kb, K, BSZ * Lkv, DM, DM);
  linear(st, Xkv, p.vw, p.vb, V, BSZ * Lkv, DM, DM);
  // scores[z,lq,lk] = (Q . K) / sqrt(DH)   (batched over B*H, headed A/B)
  gemm(st, 1, Q, DM, (long)Lq * DM, 1,
           K, DM, (long)Lkv * DM, 1,
           nullptr, SC, Lkv, (long)Lq * Lkv, 0,
           Lq, Lkv, DHEAD, 0.176776695f, 0, BSZ * NH);
  long rows = (long)BSZ * NH * Lq;
  softmax_rows_kernel<<<(unsigned)((rows + 255) / 256), 256, 0, st>>>(SC, rows, Lkv);
  // ctx = probs @ V  -> headed layout [B, Lq, 256]
  gemm(st, 0, SC, Lkv, (long)Lq * Lkv, 0,
           V, DM, (long)Lkv * DM, 1,
           nullptr, T2, DM, (long)Lq * DM, 1,
           Lq, DHEAD, Lkv, 1.0f, 0, BSZ * NH);
  linear(st, T2, p.ow, p.ob, AO, BSZ * Lq, DM, DM);
}

static void ffn_block(hipStream_t st, float* X, int rows,
                      const float* f1w, const float* f1b,
                      const float* f2w, const float* f2b,
                      const float* g, const float* b, float* T1, float* T2)
{
  linear(st, X,  f1w, f1b, T1, rows, DM, DM, 1);   // GELU
  linear(st, T1, f2w, f2b, T2, rows, DM, DM, 0);
  ln(st, T2, X, g, b, X, rows);
}

static int distil_apply(hipStream_t st, float* X, int L, const ConvP& c,
                        float* Y, float* BNM, float* BNV)
{
  int Lc = L + 2;
  // circular conv (pad 2, k=3) as three accumulated WMMA GEMMs
  for (int t = 0; t < 3; ++t) {
    gemm(st, 2, X, DM, (long)L * DM, 0,
             c.w + t, 768, 0, 0,
             (t == 0) ? c.b : nullptr,
             Y, DM, (long)Lc * DM, 0,
             Lc, DM, DM, 1.0f, (t == 0) ? 0 : 2, BSZ,
             L, t - 2);
  }
  int rowsc = BSZ * Lc;
  bn_stats_kernel<<<DM, 256, 0, st>>>(Y, rowsc, BNM, BNV);
  long n = (long)rowsc * DM;
  bn_elu_kernel<<<(unsigned)((n + 255) / 256), 256, 0, st>>>(Y, BNM, BNV, c.bg, c.bb, n);
  int Lp = (Lc - 1) / 2 + 1;
  maxpool_kernel<<<dim3((unsigned)Lp, BSZ), 256, 0, st>>>(Y, X, Lc, Lp);
  return Lp;
}

extern "C" void kernel_launch(void* const* d_in, const int* in_sizes, int n_in,
                              void* d_out, int out_size, void* d_ws, size_t ws_size,
                              hipStream_t stream)
{
  (void)in_sizes; (void)n_in; (void)out_size;
  int pi = 0;
  auto nx = [&]() { return (const float*)d_in[pi++]; };
  const float* x_enc = nx();
  const float* x_dec = nx();
  const float* enc_emb_w = nx();
  const float* dec_emb_w = nx();
  auto rd_attn = [&](AttnP& a) {
    a.qw = nx(); a.qb = nx(); a.kw = nx(); a.kb = nx();
    a.vw = nx(); a.vb = nx(); a.ow = nx(); a.ob = nx();
  };
  EncLP enc_l[3];
  for (int i = 0; i < 3; ++i) {
    rd_attn(enc_l[i].a);
    enc_l[i].g1 = nx(); enc_l[i].b1 = nx(); enc_l[i].g2 = nx(); enc_l[i].b2 = nx();
    enc_l[i].f1w = nx(); enc_l[i].f1b = nx(); enc_l[i].f2w = nx(); enc_l[i].f2b = nx();
  }
  ConvP conv_l[2];
  for (int i = 0; i < 2; ++i) {
    conv_l[i].w = nx(); conv_l[i].b = nx(); conv_l[i].bg = nx(); conv_l[i].bb = nx();
  }
  const float* enc_ng = nx(); const float* enc_nb = nx();
  DecLP dec_l[2];
  for (int j = 0; j < 2; ++j) {
    rd_attn(dec_l[j].s); rd_attn(dec_l[j].c);
    dec_l[j].g1 = nx(); dec_l[j].b1 = nx(); dec_l[j].g2 = nx(); dec_l[j].b2 = nx();
    dec_l[j].g3 = nx(); dec_l[j].b3 = nx();
    dec_l[j].f1w = nx(); dec_l[j].f1b = nx(); dec_l[j].f2w = nx(); dec_l[j].f2b = nx();
  }
  const float* dec_ng = nx(); const float* dec_nb = nx();
  const float* proj_w = nx(); const float* proj_b = nx();

  // ---- workspace ----
  size_t off = 0;
  auto wsf = [&](size_t nfl) {
    float* p = (float*)((char*)d_ws + off);
    off += ((nfl * sizeof(float) + 255) / 256) * 256;
    return p;
  };
  const size_t CAP = (size_t)BSZ * 1442 * DM;
  float* ENC = wsf(CAP);
  float* DEC = wsf(CAP);
  float* Q   = wsf(CAP);
  float* K   = wsf(CAP);
  float* V   = wsf(CAP);
  float* CTX = wsf(CAP);
  float* AO  = wsf(CAP);
  float* T1  = wsf(CAP);
  float* T2  = wsf(CAP);
  float* SC  = wsf((size_t)BSZ * NH * 720 * 362);
  float* MB  = wsf((size_t)BSZ * NH * 1442);
  int*   TOP = (int*)wsf(64 * 64);
  float* BNM = wsf(DM);
  float* BNV = wsf(DM);
  if (off > ws_size) return;

  // ---- encoder ----
  int L = 1440;
  embed_kernel<<<dim3((unsigned)L, BSZ), 256, 0, stream>>>(x_enc, enc_emb_w, ENC, L);
  for (int i = 0; i < 3; ++i) {
    int rows = BSZ * L;
    attn_prob(stream, ENC, L, enc_l[i].a, 0, 0, Q, K, V, CTX, T1, AO, MB, TOP, i);
    ln(stream, AO, ENC, enc_l[i].g1, enc_l[i].b1, ENC, rows);
    ffn_block(stream, ENC, rows, enc_l[i].f1w, enc_l[i].f1b, enc_l[i].f2w, enc_l[i].f2b,
              enc_l[i].g2, enc_l[i].b2, T1, T2);
    if (i < 2) L = distil_apply(stream, ENC, L, conv_l[i], T1, BNM, BNV);
  }
  int Lf = L;                      // 1440 -> 721 -> 362
  ln(stream, ENC, nullptr, enc_ng, enc_nb, ENC, BSZ * Lf);

  // ---- decoder ----
  const int Ld = 720;
  const int rowsD = BSZ * Ld;
  embed_kernel<<<dim3((unsigned)Ld, BSZ), 256, 0, stream>>>(x_dec, dec_emb_w, DEC, Ld);
  for (int j = 0; j < 2; ++j) {
    attn_prob(stream, DEC, Ld, dec_l[j].s, 1, 1, Q, K, V, CTX, T1, AO, MB, TOP, 100 + j);
    ln(stream, AO, DEC, dec_l[j].g1, dec_l[j].b1, DEC, rowsD);
    attn_full(stream, DEC, Ld, ENC, Lf, dec_l[j].c, Q, K, V, SC, T2, AO);
    ln(stream, AO, DEC, dec_l[j].g2, dec_l[j].b2, DEC, rowsD);
    ffn_block(stream, DEC, rowsD, dec_l[j].f1w, dec_l[j].f1b, dec_l[j].f2w, dec_l[j].f2b,
              dec_l[j].g3, dec_l[j].b3, T1, T2);
  }
  ln(stream, DEC, nullptr, dec_ng, dec_nb, DEC, rowsD);

  // ---- projection -> d_out [B, 720, 2] ----
  linear(stream, DEC, proj_w, proj_b, (float*)d_out, rowsD, DM, 2);
}
